// MetricScorer_6597069767502
// MI455X (gfx1250) — matrix operand order repaired and checked by reading, not compiled, not run on hardware
//
#include <hip/hip_runtime.h>

// ---------------------------------------------------------------------------
// L2 metric scorer for MI455X (gfx1250, wave32).
// out[b,q,k] = 1 / (1 + sqrt(max(|q|^2 + |k|^2 - 2 q.k, 1e-12)))
// Cross term via bf16x3 split (hi*hi + hi*lo + lo*hi) on v_wmma_f32_16x16x32_bf16.
//
// Fast path (needs ~168MB workspace):
//   1) l2_norm_kernel    : exact fp32 row norms
//   2) l2_convert_kernel : fp32 -> 4 bf16 planes (q_hi,q_lo,k_hi,k_lo) stored
//                          chunk-major with LDS padding baked in, so the GEMM
//                          stages tiles with raw async global->LDS B128 copies
//   3) l2_scorer_split   : WMMA GEMM, double-buffered async LDS staging,
//                          zero conversion VALU in the hot loop
// Fallback (small ws): l2_scorer_fused converts in-kernel (round-2 version).
// ---------------------------------------------------------------------------

typedef __attribute__((ext_vector_type(16))) __bf16       v16bf;
typedef __attribute__((ext_vector_type(8)))  float        v8f;
typedef __attribute__((ext_vector_type(8)))  unsigned int v8u;
typedef __attribute__((ext_vector_type(4)))  int          v4i;

#define TILE    128
#define CHUNK   32
#define NCH     16          // D_DIM / CHUNK
#define D_DIM   512
#define QK_DIM  4096
#define SH      40          // row stride in bf16 elems (32 data + 8 pad) = 80B
#define HALFB   2560        // ushorts per half tile-plane (5120B)

#if __has_builtin(__builtin_amdgcn_global_load_async_to_lds_b128) && \
    __has_builtin(__builtin_amdgcn_s_wait_asynccnt)
#define HAVE_ASYNC 1
#else
#define HAVE_ASYNC 0
#endif

typedef v4i __attribute__((address_space(1)))* gv4_ptr;   // global int4*
typedef v4i __attribute__((address_space(3)))* lv4_ptr;   // LDS int4*

union FragU { v8u u; v16bf v; };

__device__ __forceinline__ unsigned int bf16_hi_bits(float x) {
    unsigned int u = __builtin_bit_cast(unsigned int, x);
    return (u + 0x7FFFu + ((u >> 16) & 1u)) >> 16;   // RNE to bf16
}

__device__ __forceinline__ void cvt_pair(float x, float y,
                                         unsigned int& hi, unsigned int& lo) {
    unsigned int hx = bf16_hi_bits(x);
    unsigned int hy = bf16_hi_bits(y);
    float rx = x - __builtin_bit_cast(float, hx << 16);   // exact residual
    float ry = y - __builtin_bit_cast(float, hy << 16);
    unsigned int lx = bf16_hi_bits(rx);
    unsigned int ly = bf16_hi_bits(ry);
    hi = (hx & 0xFFFFu) | (hy << 16);
    lo = (lx & 0xFFFFu) | (ly << 16);
}

__device__ __forceinline__ void cvt8(float4 a, float4 b, uint4& hi, uint4& lo) {
    cvt_pair(a.x, a.y, hi.x, lo.x);
    cvt_pair(a.z, a.w, hi.y, lo.y);
    cvt_pair(b.x, b.y, hi.z, lo.z);
    cvt_pair(b.z, b.w, hi.w, lo.w);
}

// A-matrix 16x32 bf16 fragment (dense 16-bit A layout).
__device__ __forceinline__ v16bf load_a(const unsigned short* lds, int row0, int lane) {
    const int r  = row0 + (lane & 15);
    const int kh = (lane >> 4) << 3;          // 0 or 8
    const unsigned short* base = lds + r * SH;
    uint4 a = *(const uint4*)(base + kh);
    uint4 b = *(const uint4*)(base + 16 + kh);
    FragU f;
    f.u = (v8u){a.x, a.y, a.z, a.w, b.x, b.y, b.z, b.w};
    return f.v;
}

// B-matrix 32x16 bf16 fragment (sparse-B-derived dense layout).
__device__ __forceinline__ v16bf load_b(const unsigned short* lds, int col0, int lane) {
    const int c  = col0 + (lane & 15);
    const int kb = (lane >> 4) << 3;          // 0 or 8
    const unsigned short* base = lds + c * SH + kb;
    uint4 a = *(const uint4*)(base + 0);
    uint4 b = *(const uint4*)(base + 16);
    FragU f;
    f.u = (v8u){a.x, a.y, a.z, a.w, b.x, b.y, b.z, b.w};
    return f.v;
}

// bf16x3 emulated-fp32 accumulate.
__device__ __forceinline__ v8f wmma3(v16bf Ah, v16bf Al, v16bf Bh, v16bf Bl, v8f c) {
    c = __builtin_amdgcn_wmma_f32_16x16x32_bf16(false, Ah, false, Bh, (short)0, c, false, false);
    c = __builtin_amdgcn_wmma_f32_16x16x32_bf16(false, Ah, false, Bl, (short)0, c, false, false);
    c = __builtin_amdgcn_wmma_f32_16x16x32_bf16(false, Al, false, Bh, (short)0, c, false, false);
    return c;
}

// Copy 5120B (half a tile-plane) for one wave: per-lane 16B, 10 ops of 512B.
// Global/LDS layouts are byte-identical, so offsets fold into the instruction.
__device__ __forceinline__ void copy_half_plane(const unsigned short* g, unsigned short* l) {
#if HAVE_ASYNC
#pragma unroll
    for (int i = 0; i < 10; ++i)
        __builtin_amdgcn_global_load_async_to_lds_b128(
            (gv4_ptr)(unsigned long long)(g + i * 256),
            (lv4_ptr)(unsigned int)(unsigned long long)(l + i * 256),
            0, 0);
#else
#pragma unroll
    for (int i = 0; i < 10; ++i)
        *(uint4*)(l + i * 256) = *(const uint4*)(g + i * 256);
#endif
}

__device__ __forceinline__ void wait_async10() {
#if HAVE_ASYNC
    __builtin_amdgcn_s_wait_asynccnt(10);
#endif
}
__device__ __forceinline__ void wait_async0() {
#if HAVE_ASYNC
    __builtin_amdgcn_s_wait_asynccnt(0);
#endif
}

// ---------------------------------------------------------------------------
// Pass 1: exact fp32 row norms. One wave per row.
__global__ __launch_bounds__(256) void l2_norm_kernel(
    const float* __restrict__ keys, const float* __restrict__ queries,
    float* __restrict__ ksq, float* __restrict__ qsq, int nrows)
{
    const int lane = threadIdx.x & 31;
    const int row  = blockIdx.x * 8 + (threadIdx.x >> 5);
    if (row >= nrows) return;
    const float* qr = queries + (size_t)row * D_DIM;
    const float* kr = keys    + (size_t)row * D_DIM;
    float sq = 0.0f, sk = 0.0f;
#pragma unroll
    for (int p = 0; p < 4; ++p) {
        float4 a = *(const float4*)(qr + (lane << 2) + (p << 7));
        float4 c = *(const float4*)(kr + (lane << 2) + (p << 7));
        sq += a.x * a.x + a.y * a.y + a.z * a.z + a.w * a.w;
        sk += c.x * c.x + c.y * c.y + c.z * c.z + c.w * c.w;
    }
#pragma unroll
    for (int off = 16; off > 0; off >>= 1) {
        sq += __shfl_xor(sq, off, 32);
        sk += __shfl_xor(sk, off, 32);
    }
    if (lane == 0) { qsq[row] = sq; ksq[row] = sk; }
}

// ---------------------------------------------------------------------------
// Pass 2 (fast path): fp32 -> padded chunk-major bf16 planes.
// plane[c][gr][SH]: element (gr, c*32+j) at ((size_t)c*rows + gr)*SH + j.
__global__ __launch_bounds__(256) void l2_convert_kernel(
    const float* __restrict__ keys, const float* __restrict__ queries,
    unsigned short* __restrict__ qh, unsigned short* __restrict__ ql,
    unsigned short* __restrict__ kh, unsigned short* __restrict__ kl,
    int rows)
{
    const int idx = blockIdx.x * 256 + threadIdx.x;   // rows*64 threads
    const int gr  = idx >> 6;
    if (gr >= rows) return;
    const int g   = idx & 63;          // d-group of 8
    const int c   = g >> 2;
    const int dc  = (g & 3) << 3;
    const float* qp = queries + (size_t)gr * D_DIM + (c << 5) + dc;
    const float* kp = keys    + (size_t)gr * D_DIM + (c << 5) + dc;
    float4 q0 = *(const float4*)(qp);
    float4 q1 = *(const float4*)(qp + 4);
    float4 k0 = *(const float4*)(kp);
    float4 k1 = *(const float4*)(kp + 4);
    uint4 a, b2, c2, d2;
    cvt8(q0, q1, a, b2);
    cvt8(k0, k1, c2, d2);
    const size_t o = ((size_t)c * rows + gr) * SH + dc;   // 16B aligned
    *(uint4*)(qh + o) = a;
    *(uint4*)(ql + o) = b2;
    *(uint4*)(kh + o) = c2;
    *(uint4*)(kl + o) = d2;
}

// ---------------------------------------------------------------------------
// Pass 3 (fast path): WMMA GEMM with async double-buffered LDS staging.
__global__ __launch_bounds__(256) void l2_scorer_split(
    const unsigned short* __restrict__ qh, const unsigned short* __restrict__ ql,
    const unsigned short* __restrict__ kh, const unsigned short* __restrict__ kl,
    const float* __restrict__ ksq, const float* __restrict__ qsq,
    float* __restrict__ out, int rows)
{
    __shared__ __attribute__((aligned(16))) unsigned short smem[2][4][TILE * SH]; // 80KB
    __shared__ float lds_qsq[TILE];
    __shared__ float lds_ksq[TILE];

    const int t  = threadIdx.x;
    const int b  = blockIdx.z;
    const int q0 = blockIdx.y * TILE;
    const int k0 = blockIdx.x * TILE;

    if (t < TILE) {
        lds_qsq[t] = qsq[(size_t)b * QK_DIM + q0 + t];
        lds_ksq[t] = ksq[(size_t)b * QK_DIM + k0 + t];
    }

    const int lane = t & 31;
    const int wid  = t >> 5;
    const int wm   = (wid >> 1) << 5;   // wave row base: 0,32,64,96
    const int wn   = (wid & 1) << 6;    // wave col base: 0,64

    // Copy assignment: wave -> (plane, half). 4 planes x 2 halves = 8 waves.
    const int p = wid & 3;
    const int h = wid >> 2;
    const unsigned short* plane = (p == 0) ? qh : (p == 1) ? ql : (p == 2) ? kh : kl;
    const int r0 = (p < 2) ? q0 : k0;
    const unsigned short* gsrc =
        plane + ((size_t)b * QK_DIM + r0) * SH + h * HALFB + lane * 8;
    const size_t cstride = (size_t)rows * SH;       // per-chunk advance (ushorts)
    unsigned short* l0 = &smem[0][p][h * HALFB + lane * 8];
    unsigned short* l1 = &smem[1][p][h * HALFB + lane * 8];

    copy_half_plane(gsrc, l0);                      // prologue: chunk 0 -> buf 0

    v8f acc[2][4] = {};

    for (int c = 0; c < NCH; ++c) {
        if (c + 1 < NCH) {                          // prefetch next chunk
            copy_half_plane(gsrc + (size_t)(c + 1) * cstride, (c & 1) ? l0 : l1);
            wait_async10();                         // own chunk-c ops complete
        } else {
            wait_async0();
        }
        __syncthreads();                            // whole buf[c&1] valid

        const unsigned short* bqh = &smem[c & 1][0][0];
        const unsigned short* bql = &smem[c & 1][1][0];
        const unsigned short* bkh = &smem[c & 1][2][0];
        const unsigned short* bkl = &smem[c & 1][3][0];

        v16bf A0h = load_a(bqh, wm,      lane);
        v16bf A0l = load_a(bql, wm,      lane);
        v16bf A1h = load_a(bqh, wm + 16, lane);
        v16bf A1l = load_a(bql, wm + 16, lane);
#pragma unroll
        for (int j = 0; j < 4; ++j) {
            v16bf Bh = load_b(bkh, wn + 16 * j, lane);
            v16bf Bl = load_b(bkl, wn + 16 * j, lane);
            acc[0][j] = wmma3(A0h, A0l, Bh, Bl, acc[0][j]);
            acc[1][j] = wmma3(A1h, A1l, Bh, Bl, acc[1][j]);
        }
        __syncthreads();                            // reads done before overwrite
    }

    const int nl = lane & 15;
    const int mh = (lane >> 4) << 3;
#pragma unroll
    for (int i = 0; i < 2; ++i) {
#pragma unroll
        for (int j = 0; j < 4; ++j) {
            const int n  = wn + 16 * j + nl;
            const float kn = lds_ksq[n];
#pragma unroll
            for (int r = 0; r < 8; ++r) {
                const int m = wm + 16 * i + mh + r;
                float d2 = lds_qsq[m] + kn - 2.0f * acc[i][j][r];
                d2 = fmaxf(d2, 1e-12f);
                float res = 1.0f / (1.0f + sqrtf(d2));
                __builtin_nontemporal_store(
                    res, out + ((size_t)b * QK_DIM + q0 + m) * (size_t)QK_DIM + k0 + n);
            }
        }
    }
}

// ---------------------------------------------------------------------------
// Fallback: fused in-kernel conversion (used only if ws is too small).
__global__ __launch_bounds__(256) void l2_scorer_fused(
    const float* __restrict__ keys, const float* __restrict__ queries,
    const float* __restrict__ ksq, const float* __restrict__ qsq,
    float* __restrict__ out)
{
    __shared__ __attribute__((aligned(16))) unsigned short lds_qh[TILE * SH];
    __shared__ __attribute__((aligned(16))) unsigned short lds_ql[TILE * SH];
    __shared__ __attribute__((aligned(16))) unsigned short lds_kh[TILE * SH];
    __shared__ __attribute__((aligned(16))) unsigned short lds_kl[TILE * SH];
    __shared__ float lds_qsq[TILE];
    __shared__ float lds_ksq[TILE];

    const int t  = threadIdx.x;
    const int b  = blockIdx.z;
    const int q0 = blockIdx.y * TILE;
    const int k0 = blockIdx.x * TILE;

    const float* qbase = queries + ((size_t)b * QK_DIM + q0) * D_DIM;
    const float* kbase = keys    + ((size_t)b * QK_DIM + k0) * D_DIM;

    if (t < TILE) {
        lds_qsq[t] = qsq[(size_t)b * QK_DIM + q0 + t];
        lds_ksq[t] = ksq[(size_t)b * QK_DIM + k0 + t];
    }

    const int lane = t & 31;
    const int wid  = t >> 5;
    const int wm   = (wid >> 1) << 5;
    const int wn   = (wid & 1) << 6;

    v8f acc[2][4] = {};

    for (int c = 0; c < NCH; ++c) {
        const int dbase = c * CHUNK;
#pragma unroll
        for (int p = 0; p < 2; ++p) {
            const int g  = t + p * 256;
            const int r  = g >> 2;
            const int dc = (g & 3) << 3;
            const float* qp = qbase + (size_t)r * D_DIM + dbase + dc;
            const float* kp = kbase + (size_t)r * D_DIM + dbase + dc;
            float4 qv0 = *(const float4*)(qp);
            float4 qv1 = *(const float4*)(qp + 4);
            float4 kv0 = *(const float4*)(kp);
            float4 kv1 = *(const float4*)(kp + 4);
            uint4 a, b2, c2, d2;
            cvt8(qv0, qv1, a, b2);
            cvt8(kv0, kv1, c2, d2);
            const int o = r * SH + dc;
            *(uint4*)(lds_qh + o) = a;
            *(uint4*)(lds_ql + o) = b2;
            *(uint4*)(lds_kh + o) = c2;
            *(uint4*)(lds_kl + o) = d2;
        }
        __syncthreads();

        v16bf A0h = load_a(lds_qh, wm,      lane);
        v16bf A0l = load_a(lds_ql, wm,      lane);
        v16bf A1h = load_a(lds_qh, wm + 16, lane);
        v16bf A1l = load_a(lds_ql, wm + 16, lane);
#pragma unroll
        for (int j = 0; j < 4; ++j) {
            v16bf Bh = load_b(lds_kh, wn + 16 * j, lane);
            v16bf Bl = load_b(lds_kl, wn + 16 * j, lane);
            acc[0][j] = wmma3(A0h, A0l, Bh, Bl, acc[0][j]);
            acc[1][j] = wmma3(A1h, A1l, Bh, Bl, acc[1][j]);
        }
        __syncthreads();
    }

    const int nl = lane & 15;
    const int mh = (lane >> 4) << 3;
#pragma unroll
    for (int i = 0; i < 2; ++i) {
#pragma unroll
        for (int j = 0; j < 4; ++j) {
            const int n  = wn + 16 * j + nl;
            const float kn = lds_ksq[n];
#pragma unroll
            for (int r = 0; r < 8; ++r) {
                const int m = wm + 16 * i + mh + r;
                float d2 = lds_qsq[m] + kn - 2.0f * acc[i][j][r];
                d2 = fmaxf(d2, 1e-12f);
                float res = 1.0f / (1.0f + sqrtf(d2));
                __builtin_nontemporal_store(
                    res, out + ((size_t)b * QK_DIM + q0 + m) * (size_t)QK_DIM + k0 + n);
            }
        }
    }
}

// ---------------------------------------------------------------------------
extern "C" void kernel_launch(void* const* d_in, const int* in_sizes, int n_in,
                              void* d_out, int out_size, void* d_ws, size_t ws_size,
                              hipStream_t stream) {
    (void)n_in; (void)out_size;
    const float* keys    = (const float*)d_in[0];   // setup_inputs order: keys, queries
    const float* queries = (const float*)d_in[1];
    float* out = (float*)d_out;

    const int B    = in_sizes[0] / (QK_DIM * D_DIM);
    const int rows = B * QK_DIM;

    float* ksq = (float*)d_ws;
    float* qsq = ksq + rows;

    const size_t planeElems = (size_t)NCH * rows * SH;              // ushorts
    const size_t need = (size_t)2 * rows * sizeof(float)
                      + 4 * planeElems * sizeof(unsigned short);    // ~168MB

    l2_norm_kernel<<<dim3((rows + 7) / 8), dim3(256), 0, stream>>>(
        keys, queries, ksq, qsq, rows);

    dim3 grid(QK_DIM / TILE, QK_DIM / TILE, B);

    if (ws_size >= need) {
        unsigned short* qh = (unsigned short*)(qsq + rows);
        unsigned short* ql = qh + planeElems;
        unsigned short* kh = ql + planeElems;
        unsigned short* kl = kh + planeElems;
        l2_convert_kernel<<<dim3(rows / 4), dim3(256), 0, stream>>>(
            keys, queries, qh, ql, kh, kl, rows);
        l2_scorer_split<<<grid, dim3(256), 0, stream>>>(
            qh, ql, kh, kl, ksq, qsq, out, rows);
    } else {
        l2_scorer_fused<<<grid, dim3(256), 0, stream>>>(
            keys, queries, ksq, qsq, out);
    }
}
